// TnoCausalV1_2508260900872
// MI455X (gfx1250) — matrix operand order repaired and checked by reading, not compiled, not run on hardware
//
#include <hip/hip_runtime.h>
#include <hip/hip_bf16.h>

#define BATCH 8
#define SEQ   4096
#define DIM   1024
#define TOFS  112   // tt[d][s] = t[s-TOFS][d], zero outside [0,SEQ): covers 4-tile tails
#define TS    4256  // >= SEQ + TOFS + 48, multiple of 32

#define XT_BYTES  ((size_t)DIM * BATCH * SEQ * 2)   // 64 MiB
#define TT_BYTES  ((size_t)DIM * TS * 2)            // ~8.3 MiB
#define ZPAD_U32  64                                // 256-byte zero pad for invalid A rows

typedef __attribute__((ext_vector_type(16))) __bf16        v16bf;
typedef __attribute__((ext_vector_type(8)))  float         v8f;
typedef __attribute__((ext_vector_type(4)))  unsigned int  u32x4;
typedef __attribute__((ext_vector_type(2)))  unsigned int  u32x2;

union FragAB {
  u32x4          q[2];
  unsigned short u[16];
  v16bf          v;
};

// ---- kernel 0: zero the pad region used by padded A rows (batch 8..15)
__global__ void zclear_kernel(unsigned int* __restrict__ p) {
  p[threadIdx.x] = 0u;
}

// ---- kernel 1: x (B,N,D) f32  ->  xt (D,B,N) bf16  (channel-major, k contiguous)
__global__ void xpose_kernel(const float* __restrict__ x,
                             __hip_bfloat16* __restrict__ xt) {
  __shared__ float tile[32][33];
  const int c  = blockIdx.z;
  const int k0 = blockIdx.x * 32;
  const int d0 = blockIdx.y * 32;
  const int tx = threadIdx.x & 31;
  const int ty = threadIdx.x >> 5;   // 0..7
#pragma unroll
  for (int r = ty; r < 32; r += 8)
    tile[r][tx] = x[((size_t)c * SEQ + k0 + r) * DIM + d0 + tx];
  __syncthreads();
#pragma unroll
  for (int r = ty; r < 32; r += 8)
    xt[((size_t)(d0 + r) * BATCH + c) * SEQ + k0 + tx] =
        __float2bfloat16(tile[tx][r]);
}

// ---- kernel 2: t (N,D) f32 -> tt (D,TS) bf16, shifted by TOFS, zero-padded
__global__ void tpose_kernel(const float* __restrict__ t,
                             __hip_bfloat16* __restrict__ tt) {
  __shared__ float tile[32][33];
  const int s0 = blockIdx.x * 32;   // TS/32 = 133 blocks
  const int d0 = blockIdx.y * 32;
  const int tx = threadIdx.x & 31;
  const int ty = threadIdx.x >> 5;
#pragma unroll
  for (int r = ty; r < 32; r += 8) {
    const int src = s0 + r - TOFS;
    tile[r][tx] = (src >= 0 && src < SEQ) ? t[(size_t)src * DIM + d0 + tx] : 0.0f;
  }
  __syncthreads();
#pragma unroll
  for (int r = ty; r < 32; r += 8)
    tt[(size_t)(d0 + r) * TS + s0 + tx] = __float2bfloat16(tile[tx][r]);
}

// ---- kernel 3: main Toeplitz WMMA kernel
// Block: 512 threads = 16 waves; wave w handles channel d0+w and FOUR i-tiles
// [i_base + 16j, j=0..3]. Per K-chunk (32): one shared A fragment (x batch rows)
// feeds 4 independent wmmas against 4 shifted Toeplitz B fragments.
__global__ __launch_bounds__(512)
void toeplitz_wmma_kernel(const __hip_bfloat16* __restrict__ xt_,
                          const __hip_bfloat16* __restrict__ tt_,
                          const unsigned int* __restrict__ zpad,
                          float* __restrict__ y) {
  __shared__ __align__(16) unsigned short tw[16 * 128];  // per-wave t-window (256B each)
  __shared__ float oy[8192];                             // (c, i(64), d_local(16)) tile

  const unsigned short* xt   = (const unsigned short*)xt_;
  const unsigned int*   tt32 = (const unsigned int*)tt_;

  const int tid    = threadIdx.x;
  const int lane   = tid & 31;
  const int wave   = tid >> 5;            // d_local 0..15
  const int i_base = blockIdx.x * 64;
  const int d0     = blockIdx.y * 16;
  const int d      = d0 + wave;

  const int  cc     = lane & 15;          // A row = batch index (valid < 8)
  const int  half   = lane >> 4;          // ISA 16-bit A layout half-select
  const bool cvalid = cc < 8;

  // Per-lane A pointer: valid lanes walk their x row; invalid lanes camp on a
  // zeroed 256B pad with stride 0 -> all loads unconditional, no exec masking.
  const unsigned short* ap =
      cvalid ? (xt + ((size_t)d * BATCH + cc) * SEQ + half * 16)
             : ((const unsigned short*)zpad + half * 16);
  const ptrdiff_t astep = cvalid ? 32 : 0;              // ushorts per K-chunk

  // Per-lane t-window staging pointer (b64 per lane covers 128 ushorts/wave).
  // Window start s0 = i_base - kb + TOFS - 32; slides down 32 ushorts per chunk.
  const u32x2* tp = (const u32x2*)(tt32 + (size_t)d * (TS / 2) +
                                   ((i_base + TOFS - 32) >> 1)) + lane;

  // Fixed LDS addresses.
  u32x2*                tws = (u32x2*)tw + wave * 32 + lane;   // staging store (b64)
  const unsigned short* twb = tw + wave * 128 + 32 - cc + 16 * half;  // B-frag base, +16j per tile

  v8f acc0 = {}, acc1 = {}, acc2 = {}, acc3 = {};
  const int nchunks = (i_base >> 5) + 2;   // kb = 0,32,... while kb <= i_base+63

#pragma unroll 2
  for (int it = nchunks; it > 0; --it) {
    // Stage this wave's 95-value t-window into LDS: one b64 per lane.
    *tws = *tp;
    __builtin_prefetch(tp - 8, 0, 1);                   // next (lower) t-window
    asm volatile("" ::: "memory");  // keep ds_store before ds_loads (same-wave LDS in-order)

    // Shared A fragment: elements 0..15 = x[c, kb+16h..kb+16h+15]
    FragAB a;
    a.q[0] = *(const u32x4*)(ap);
    a.q[1] = *(const u32x4*)(ap + 8);
    __builtin_prefetch(ap + 32, 0, 1);                  // next K-chunk's A stream

    // Four B fragments: tile j reads window offsets 32+16j+16h+n-cc (n=0..15).
    FragAB b0, b1, b2, b3;
#pragma unroll
    for (int n = 0; n < 16; ++n) b0.u[n] = twb[n];
#pragma unroll
    for (int n = 0; n < 16; ++n) b1.u[n] = twb[16 + n];
#pragma unroll
    for (int n = 0; n < 16; ++n) b2.u[n] = twb[32 + n];
#pragma unroll
    for (int n = 0; n < 16; ++n) b3.u[n] = twb[48 + n];

    // Four independent wmmas -> back-to-back XDL issue.
    acc0 = __builtin_amdgcn_wmma_f32_16x16x32_bf16(false, a.v, false, b0.v,
                                                   (short)0, acc0, false, false);
    acc1 = __builtin_amdgcn_wmma_f32_16x16x32_bf16(false, a.v, false, b1.v,
                                                   (short)0, acc1, false, false);
    acc2 = __builtin_amdgcn_wmma_f32_16x16x32_bf16(false, a.v, false, b2.v,
                                                   (short)0, acc2, false, false);
    acc3 = __builtin_amdgcn_wmma_f32_16x16x32_bf16(false, a.v, false, b3.v,
                                                   (short)0, acc3, false, false);
    asm volatile("" ::: "memory");  // WAR: next staging store after this iter's ds reads

    ap += astep;
    tp -= 8;
  }

  // D layout: lane l<16 (h=0), VGPR v: value = y[batch=v, i_base + 16j + l, d].
  if (lane < 16) {
#pragma unroll
    for (int v = 0; v < 8; ++v) {
      oy[v * 1024 + (lane)      * 16 + wave] = acc0[v];
      oy[v * 1024 + (16 + lane) * 16 + wave] = acc1[v];
      oy[v * 1024 + (32 + lane) * 16 + wave] = acc2[v];
      oy[v * 1024 + (48 + lane) * 16 + wave] = acc3[v];
    }
  }
  __syncthreads();

  // Coalesced stores: 16 consecutive d per 64B segment.
#pragma unroll
  for (int it = 0; it < 16; ++it) {
    const int e  = tid + it * 512;
    const int c  = e >> 10;
    const int ii = (e >> 4) & 63;
    const int dl = e & 15;
    y[((size_t)c * SEQ + i_base + ii) * DIM + d0 + dl] = oy[e];
  }
}

extern "C" void kernel_launch(void* const* d_in, const int* in_sizes, int n_in,
                              void* d_out, int out_size, void* d_ws, size_t ws_size,
                              hipStream_t stream) {
  const float* x = (const float*)d_in[0];   // (8, 4096, 1024) f32
  const float* t = (const float*)d_in[1];   // (4096, 1024) f32
  float* y = (float*)d_out;                 // (8, 4096, 1024) f32

  // Workspace layout: xt bf16 (D,B,N) | tt bf16 (D,TS) | 256B zero pad
  __hip_bfloat16* xt   = (__hip_bfloat16*)d_ws;
  __hip_bfloat16* tt   = (__hip_bfloat16*)((char*)d_ws + XT_BYTES);
  unsigned int*   zpad = (unsigned int*)((char*)d_ws + XT_BYTES + TT_BYTES);

  zclear_kernel<<<1, ZPAD_U32, 0, stream>>>(zpad);
  xpose_kernel<<<dim3(SEQ / 32, DIM / 32, BATCH), 256, 0, stream>>>(x, xt);
  tpose_kernel<<<dim3(TS / 32, DIM / 32), 256, 0, stream>>>(t, tt);
  toeplitz_wmma_kernel<<<dim3(SEQ / 64, DIM / 16), 512, 0, stream>>>(xt, tt, zpad, y);
}
